// NormalizedCorrelation2D_58798102282739
// MI455X (gfx1250) — matrix operand order repaired and checked by reading, not compiled, not run on hardware
//
#include <hip/hip_runtime.h>
#include <hip/hip_bf16.h>

typedef __attribute__((ext_vector_type(8)))  _Float16 v8h;
typedef __attribute__((ext_vector_type(16))) _Float16 v16h;
typedef __attribute__((ext_vector_type(8)))  float    v8f;

#define NB    32            // batch
#define WW    900           // width
#define CC    256           // channels
#define MP    1024          // padded M/N (8 x 128)
#define BT    128           // block tile (M and N)
#define NTB   (MP / BT)     // 8
#define LDSW  40            // LDS row stride in halves (80B, 16B aligned, bank-spread)

// ---------------- kernel 0: zero output + norm accumulators ----------------
__global__ void nc_zero_kernel(float* __restrict__ out, int nOut,
                               float* __restrict__ sums, int nSums) {
    int i = blockIdx.x * blockDim.x + threadIdx.x;
    if (i < nOut)  out[i]  = 0.0f;
    if (i < nSums) sums[i] = 0.0f;
}

// ---------------- kernel 1: per-(b,c) sum of squares along W ----------------
// grid (NB, 9), block CC; each block covers 100 consecutive w for all channels
__global__ void nc_sums_kernel(const float* __restrict__ x1,
                               const float* __restrict__ x2,
                               float* __restrict__ s1, float* __restrict__ s2) {
    const int b  = blockIdx.x;
    const int wc = blockIdx.y;
    const int c  = threadIdx.x;
    const int w0 = wc * 100;
    const float* p1 = x1 + ((size_t)b * WW + w0) * CC + c;
    const float* p2 = x2 + ((size_t)b * WW + w0) * CC + c;
    float a1 = 0.f, a2 = 0.f;
    for (int w = 0; w < 100; ++w) {
        float u = p1[(size_t)w * CC]; a1 += u * u;
        float v = p2[(size_t)w * CC]; a2 += v * v;
    }
    atomicAdd(&s1[b * CC + c], a1);
    atomicAdd(&s2[b * CC + c], a2);
}

// ---------------- kernel 2: scale-fold + f32 -> f16 convert, pad to MP rows ----------------
// grid (MP, NB), block CC
__global__ void nc_convert_kernel(const float* __restrict__ x1,
                                  const float* __restrict__ x2,
                                  const float* __restrict__ s1,
                                  const float* __restrict__ s2,
                                  _Float16* __restrict__ A16,
                                  _Float16* __restrict__ B16) {
    const int w = blockIdx.x;
    const int b = blockIdx.y;
    const int c = threadIdx.x;
    const float scale = rsqrtf(fmaxf(s1[b * CC + c], 1e-12f)) *
                        rsqrtf(fmaxf(s2[b * CC + c], 1e-12f));
    float a = 0.f, v = 0.f;
    if (w < WW) {
        size_t idx = ((size_t)b * WW + w) * CC + c;
        a = x1[idx];
        v = x2[idx] * scale;   // both L2 norms folded into the x2 operand
    }
    size_t o = ((size_t)b * MP + w) * CC + c;
    A16[o] = (_Float16)a;
    B16[o] = (_Float16)v;
}

// ---------------- kernel 3: per-batch Gram GEMM (LDS-staged) + diagonal reduce ----------------
// grid (NTB, NTB, NB), block 256 (8 waves); block tile 128x128, wave tile 64x32 (4x2 WMMA)
__global__ __launch_bounds__(256)
void nc_corr_wmma_kernel(const _Float16* __restrict__ A16,
                         const _Float16* __restrict__ B16,
                         float* __restrict__ out) {
    __shared__ _Float16 Alds[BT * LDSW];   // 128 rows x 32 k-halves (stride 40)
    __shared__ _Float16 Blds[BT * LDSW];
    __shared__ float    diag[256];         // 255 wrapped diagonals of the 128x128 tile

    const int b     = blockIdx.z;
    const int baseM = blockIdx.y * BT;
    const int baseN = blockIdx.x * BT;
    const int tid   = threadIdx.x;
    const int wid   = tid >> 5;
    const int lane  = tid & 31;
    const int lm    = lane & 15;       // row/col within a 16-tile
    const int h     = lane >> 4;       // K-half select
    const int wr    = (wid >> 2) * 64; // wave row base within block tile (0 or 64)
    const int wcn   = (wid & 3) * 32;  // wave col base within block tile (0,32,64,96)

    diag[tid] = 0.0f;

    const _Float16* Ab = A16 + (size_t)b * MP * CC;
    const _Float16* Bb = B16 + (size_t)b * MP * CC;

    v8f acc[4][2] = {};

    for (int k0 = 0; k0 < CC; k0 += 32) {
        __syncthreads();   // previous compute (and initial diag zero) complete
        // Cooperative stage: 128 rows x 32 halves for A and B.
        // 512 chunks of 8 halves each; thread t handles chunks t and t+256.
        #pragma unroll
        for (int rep = 0; rep < 2; ++rep) {
            const int chunk = tid + rep * 256;
            const int row   = chunk >> 2;            // 0..127
            const int part  = (chunk & 3) * 8;       // 0,8,16,24 halves
            v8h av = *(const v8h*)(Ab + (size_t)(baseM + row) * CC + k0 + part);
            v8h bv = *(const v8h*)(Bb + (size_t)(baseN + row) * CC + k0 + part);
            *(v8h*)(&Alds[row * LDSW + part]) = av;
            *(v8h*)(&Blds[row * LDSW + part]) = bv;
        }
        __syncthreads();

        // Fragment loads from LDS.
        // A layout (16-bit A 16x32): lane holds chunks at k = h*8 and 16+h*8.
        v16h af[4];
        #pragma unroll
        for (int ti = 0; ti < 4; ++ti) {
            const int rb = (wr + ti * 16 + lm) * LDSW + h * 8;
            af[ti] = __builtin_shufflevector(*(const v8h*)(&Alds[rb]),
                                             *(const v8h*)(&Alds[rb + 16]),
                                             0,1,2,3,4,5,6,7,8,9,10,11,12,13,14,15);
        }
        // B layout: lane holds chunks at k = h*8 and 16+h*8 (same as A).
        v16h bf[2];
        #pragma unroll
        for (int tj = 0; tj < 2; ++tj) {
            const int nb = (wcn + tj * 16 + lm) * LDSW + h * 8;
            bf[tj] = __builtin_shufflevector(*(const v8h*)(&Blds[nb]),
                                             *(const v8h*)(&Blds[nb + 16]),
                                             0,1,2,3,4,5,6,7,8,9,10,11,12,13,14,15);
        }
        #pragma unroll
        for (int ti = 0; ti < 4; ++ti)
            #pragma unroll
            for (int tj = 0; tj < 2; ++tj)
                acc[ti][tj] = __builtin_amdgcn_wmma_f32_16x16x32_f16(
                    false, af[ti], false, bf[tj], (short)0, acc[ti][tj], false, false);
    }

    // Epilogue: G[gr,gi] contributes to out[b, (gr-gi-450) mod 900].
    // Within this 128x128 tile, (gr-gi)-(baseM-baseN) in [-127,127] -> 255 LDS bins.
    // C/D layout: lane holds (m = r + 8*h, n = lm) for r = 0..7.
    #pragma unroll
    for (int ti = 0; ti < 4; ++ti) {
        #pragma unroll
        for (int tj = 0; tj < 2; ++tj) {
            const int mloc = wr + ti * 16 + 8 * h;
            const int nloc = wcn + tj * 16 + lm;
            #pragma unroll
            for (int r = 0; r < 8; ++r)
                atomicAdd(&diag[(mloc + r) - nloc + 127], acc[ti][tj][r]);
        }
    }
    __syncthreads();

    if (tid < 255) {
        float v = diag[tid];
        if (v != 0.0f) {
            int grgi = (baseM - baseN) + tid - 127;  // gr - gi
            int j = grgi - 450;
            j %= WW; if (j < 0) j += WW;
            atomicAdd(&out[b * WW + j], v);
        }
    }
}

extern "C" void kernel_launch(void* const* d_in, const int* in_sizes, int n_in,
                              void* d_out, int out_size, void* d_ws, size_t ws_size,
                              hipStream_t stream) {
    const float* x1 = (const float*)d_in[0];
    const float* x2 = (const float*)d_in[1];
    float* out = (float*)d_out;

    char* ws = (char*)d_ws;
    float* s1 = (float*)ws;                         // NB*CC floats
    float* s2 = s1 + NB * CC;                       // NB*CC floats
    _Float16* A16 = (_Float16*)(ws + 65536);        // NB*MP*CC halves (16 MB)
    _Float16* B16 = A16 + (size_t)NB * MP * CC;     // NB*MP*CC halves (16 MB)

    const int nOut  = NB * WW;       // 28800
    const int nSums = 2 * NB * CC;   // 16384 (s1 and s2 are contiguous)

    nc_zero_kernel<<<(nOut + 255) / 256, 256, 0, stream>>>(out, nOut, s1, nSums);
    nc_sums_kernel<<<dim3(NB, 9), CC, 0, stream>>>(x1, x2, s1, s2);
    nc_convert_kernel<<<dim3(MP, NB), CC, 0, stream>>>(x1, x2, s1, s2, A16, B16);
    nc_corr_wmma_kernel<<<dim3(NTB, NTB, NB), 256, 0, stream>>>(A16, B16, out);
}
